// DIEN_38646115729852
// MI455X (gfx1250) — compile-verified, matrix operand order checked by reading, not run on hardware
//
#include <hip/hip_runtime.h>
#include <hip/hip_bf16.h>
#include <math.h>

// ---------------- constants ----------------
constexpr int kB = 8192;
constexpr int kT = 50;
constexpr int kAuxRows = kB * (kT - 1);   // 401408, divisible by 16
constexpr int kAttRows = kB * kT;         // 409600, divisible by 16

typedef __attribute__((ext_vector_type(16))) _Float16 v16h;
typedef __attribute__((ext_vector_type(8)))  float    v8f;

__device__ __forceinline__ float sigf(float x) { return 1.0f / (1.0f + __expf(-x)); }

// ---------------- WMMA helpers (wave32, v_wmma_f32_16x16x32_f16) ----------------
// A fragment: 16x32 f16 tile, row-major in LDS with stride lda.
// lane L: row = L&15 ; element e -> K = ((e>>3)<<4) + ((L>>4)<<3) + (e&7)
__device__ __forceinline__ v16h frag_a(const _Float16* in, int lda, int lane, int k0) {
  v16h a;
  const _Float16* p = in + (lane & 15) * lda + k0 + ((lane >> 4) << 3);
#pragma unroll
  for (int e = 0; e < 16; ++e) a[e] = p[((e >> 3) << 4) + (e & 7)];
  return a;
}
// B fragment: 32x16 f16 tile from weights [Kp][S] row-major (stride ldb), cols n0..n0+15.
__device__ __forceinline__ v16h frag_b(const _Float16* w, int ldb, int lane, int k0, int n0) {
  v16h b;
  const _Float16* p = w + (k0 + ((lane >> 4) << 3)) * ldb + n0 + (lane & 15);
#pragma unroll
  for (int e = 0; e < 16; ++e) b[e] = p[(((e >> 3) << 4) + (e & 7)) * ldb];
  return b;
}
// One 16x16 output tile (cols n0..n0+15), bias-initialized accumulator, K loop in steps of 32.
__device__ __forceinline__ v8f wmma_tile(const _Float16* in, int Kp, const _Float16* w, int ldb,
                                         const float* bias, int N, int n0, int lane) {
  int col = n0 + (lane & 15);
  float bc = (col < N) ? bias[col] : 0.0f;
  v8f acc;
#pragma unroll
  for (int r = 0; r < 8; ++r) acc[r] = bc;
  for (int k0 = 0; k0 < Kp; k0 += 32) {
    v16h a = frag_a(in, Kp, lane, k0);
    v16h b = frag_b(w, ldb, lane, k0, n0);
    acc = __builtin_amdgcn_wmma_f32_16x16x32_f16(false, a, false, b, (short)0, acc, false, false);
  }
  return acc;
}
// Full FC layer for one 16-row tile: out[16][S] = act(in @ W + bias), padded cols forced to 0.
template <int ACT>  // 0 none, 1 sigmoid, 2 relu
__device__ __forceinline__ void fc_layer(const _Float16* in, int Kp, const _Float16* w, int S,
                                         const float* bias, int N, _Float16* out, int lane) {
  for (int n0 = 0; n0 < S; n0 += 16) {
    v8f acc = wmma_tile(in, Kp, w, S, bias, N, n0, lane);
    int col = n0 + (lane & 15);
    int rb = (lane >> 4) << 3;
#pragma unroll
    for (int r = 0; r < 8; ++r) {
      float v = acc[r];
      if (ACT == 1) v = sigf(v);
      if (ACT == 2) v = fmaxf(v, 0.0f);
      out[(rb + r) * S + col] = (col < N) ? (_Float16)v : (_Float16)0.0f;
    }
  }
}
// Cooperative f32->f16 weight load into LDS with zero padding ([K][N] -> [Kp][S]).
__device__ __forceinline__ void load_weights(const float* W, int K, int N, _Float16* w, int Kp,
                                             int S, int tid, int nt) {
  for (int i = tid; i < Kp * S; i += nt) {
    int k = i / S, n = i - k * S;
    w[i] = (k < K && n < N) ? (_Float16)W[k * N + n] : (_Float16)0.0f;
  }
}

// ---------------- kernels ----------------
__global__ void k_init(float* p, int n) {
  int i = blockIdx.x * blockDim.x + threadIdx.x;
  if (i < n) p[i] = 0.0f;
}

// history / noclk embedding gather, one thread per (b,t)
__global__ void k_embed(const int* hi, const int* hc, const int* ni, const int* nc,
                        const float* ei, const float* ec, float* his, float* noclk) {
  int i = blockIdx.x * blockDim.x + threadIdx.x;
  if (i >= kB * kT) return;
  const float* a = ei + hi[i] * 4;
  const float* b = ec + hc[i] * 4;
  const float* a2 = ei + ni[i] * 4;
  const float* b2 = ec + nc[i] * 4;
  float* o = his + (size_t)i * 8;
  float* o2 = noclk + (size_t)i * 8;
#pragma unroll
  for (int e = 0; e < 4; ++e) { o[e] = a[e]; o[4 + e] = b[e]; o2[e] = a2[e]; o2[4 + e] = b2[e]; }
}

// per-batch: uid/item embeddings, history sum, attention query (PReLU)
__global__ void k_embed_b(const int* UID, const int* ITEM, const int* CAT,
                          const float* eu, const float* ei, const float* ec, const float* his,
                          const float* qw, const float* qb, const float* prelu,
                          float* uid_e, float* item_e, float* hsum, float* q) {
  int b = blockIdx.x * blockDim.x + threadIdx.x;
  if (b >= kB) return;
#pragma unroll
  for (int e = 0; e < 4; ++e) uid_e[b * 4 + e] = eu[UID[b] * 4 + e];
  float it[8];
#pragma unroll
  for (int e = 0; e < 4; ++e) { it[e] = ei[ITEM[b] * 4 + e]; it[4 + e] = ec[CAT[b] * 4 + e]; }
#pragma unroll
  for (int e = 0; e < 8; ++e) item_e[b * 8 + e] = it[e];
  float hs[8];
#pragma unroll
  for (int e = 0; e < 8; ++e) hs[e] = 0.0f;
  for (int t = 0; t < kT; ++t)
#pragma unroll
    for (int e = 0; e < 8; ++e) hs[e] += his[((size_t)(b * kT + t)) * 8 + e];
#pragma unroll
  for (int e = 0; e < 8; ++e) hsum[b * 8 + e] = hs[e];
  float pr = prelu[0];
#pragma unroll
  for (int n = 0; n < 8; ++n) {
    float v = qb[n];
#pragma unroll
    for (int k = 0; k < 8; ++k) v += it[k] * qw[k * 8 + n];
    q[b * 8 + n] = (v > 0.0f) ? v : pr * v;
  }
}

// GRU1 scan, one thread per row, weights in LDS. Output masked by seq length.
__global__ void k_gru1(const float* his, const int* seqlen, const float* wih, const float* whh,
                       const float* bih, const float* bhh, float* rnn1) {
  __shared__ float s_wih[24 * 8], s_whh[24 * 8], s_bih[24], s_bhh[24];
  int tid = threadIdx.x;
  for (int i = tid; i < 24 * 8; i += blockDim.x) { s_wih[i] = wih[i]; s_whh[i] = whh[i]; }
  for (int i = tid; i < 24; i += blockDim.x) { s_bih[i] = bih[i]; s_bhh[i] = bhh[i]; }
  __syncthreads();
  int b = blockIdx.x * blockDim.x + tid;
  if (b >= kB) return;
  int L = seqlen[b];
  float h[8];
#pragma unroll
  for (int u = 0; u < 8; ++u) h[u] = 0.0f;
  for (int t = 0; t < kT; ++t) {
    const float* x = his + ((size_t)(b * kT + t)) * 8;
    float gi[24], gh[24];
#pragma unroll
    for (int j = 0; j < 24; ++j) {
      float a = s_bih[j], c = s_bhh[j];
#pragma unroll
      for (int k = 0; k < 8; ++k) { a += x[k] * s_wih[j * 8 + k]; c += h[k] * s_whh[j * 8 + k]; }
      gi[j] = a; gh[j] = c;
    }
    float hn[8];
#pragma unroll
    for (int u = 0; u < 8; ++u) {
      float r = sigf(gi[u] + gh[u]);
      float z = sigf(gi[8 + u] + gh[8 + u]);
      float n = tanhf(gi[16 + u] + r * gh[16 + u]);
      hn[u] = (1.0f - z) * n + z * h[u];
    }
#pragma unroll
    for (int u = 0; u < 8; ++u) {
      h[u] = hn[u];
      rnn1[((size_t)(b * kT + t)) * 8 + u] = (t < L) ? h[u] : 0.0f;
    }
  }
}

// BN batch statistics for aux inputs: 24 features (rnn1 0-7 shared, his 8-15, noclk 16-23)
__global__ void k_stats24(const float* rnn1, const float* his, const float* noclk, float* sums,
                          float* sqs) {
  __shared__ float ss[24], sq[24];
  if (threadIdx.x < 24) { ss[threadIdx.x] = 0.0f; sq[threadIdx.x] = 0.0f; }
  __syncthreads();
  int i = blockIdx.x * blockDim.x + threadIdx.x;
  if (i < kAuxRows) {
    int b = i / (kT - 1), t = i % (kT - 1);
    float f[24];
#pragma unroll
    for (int e = 0; e < 8; ++e) {
      f[e] = rnn1[((size_t)(b * kT + t)) * 8 + e];
      f[8 + e] = his[((size_t)(b * kT + t + 1)) * 8 + e];
      f[16 + e] = noclk[((size_t)(b * kT + t + 1)) * 8 + e];
    }
#pragma unroll
    for (int j = 0; j < 24; ++j) { atomicAdd(&ss[j], f[j]); atomicAdd(&sq[j], f[j] * f[j]); }
  }
  __syncthreads();
  if (threadIdx.x < 24) { atomicAdd(&sums[threadIdx.x], ss[threadIdx.x]); atomicAdd(&sqs[threadIdx.x], sq[threadIdx.x]); }
}

// Auxiliary net (WMMA): BN -> 16->100->50->2 (no activations), emit channel-0 logit per row.
__global__ __launch_bounds__(128) void k_aux(const float* rnn1, const float* hisx,
                                             const float* sums, const float* sqs, int boff,
                                             const float* bng, const float* bnb, const float* W1,
                                             const float* b1, const float* W2, const float* b2,
                                             const float* W3, const float* b3, float* y0) {
  __shared__ _Float16 sW1[32 * 128], sW2[128 * 64], sW3[64 * 16];
  __shared__ _Float16 sIn[4][16 * 32], sH1[4][16 * 128], sH2[4][16 * 64];
  int tid = threadIdx.x, nt = blockDim.x;
  load_weights(W1, 16, 100, sW1, 32, 128, tid, nt);
  load_weights(W2, 100, 50, sW2, 128, 64, tid, nt);
  load_weights(W3, 50, 2, sW3, 64, 16, tid, nt);
  __syncthreads();
  int lane = tid & 31, w = tid >> 5;
  int wave = blockIdx.x * 4 + w, nw = gridDim.x * 4;
  const float invN = 1.0f / (float)kAuxRows;
  for (int tile = wave; tile < kAuxRows / 16; tile += nw) {
    _Float16* in = sIn[w];
    for (int idx = lane; idx < 16 * 32; idx += 32) {
      int row = idx >> 5, col = idx & 31;
      float v = 0.0f;
      if (col < 16) {
        int rg = tile * 16 + row;
        int b = rg / (kT - 1), t = rg % (kT - 1);
        float x = (col < 8) ? rnn1[((size_t)(b * kT + t)) * 8 + col]
                            : hisx[((size_t)(b * kT + t + 1)) * 8 + (col - 8)];
        int j = (col < 8) ? col : (boff + col - 8);
        float m = sums[j] * invN;
        float var = sqs[j] * invN - m * m;
        v = (x - m) * rsqrtf(var + 1e-5f) * bng[col] + bnb[col];
      }
      in[idx] = (_Float16)v;
    }
    fc_layer<0>(in, 32, sW1, 128, b1, 100, sH1[w], lane);
    fc_layer<0>(sH1[w], 128, sW2, 64, b2, 50, sH2[w], lane);
    v8f acc = wmma_tile(sH2[w], 64, sW3, 16, b3, 2, 0, lane);
    if ((lane & 15) == 0) {
      int rb = (lane >> 4) << 3;
#pragma unroll
      for (int r = 0; r < 8; ++r) y0[tile * 16 + rb + r] = acc[r];
    }
  }
}

// softmax-over-time (per channel-0) + aux loss accumulation
__global__ void k_aux_loss(const float* y0c, const float* y0n, const int* seqlen, float* loss_acc) {
  int b = blockIdx.x * blockDim.x + threadIdx.x;
  if (b >= kB) return;
  int L = seqlen[b];
  float mc = -1e30f, mn = -1e30f;
  for (int t = 0; t < kT - 1; ++t) {
    mc = fmaxf(mc, y0c[b * (kT - 1) + t]);
    mn = fmaxf(mn, y0n[b * (kT - 1) + t]);
  }
  float sc = 0.0f, sn = 0.0f;
  for (int t = 0; t < kT - 1; ++t) {
    sc += __expf(y0c[b * (kT - 1) + t] - mc);
    sn += __expf(y0n[b * (kT - 1) + t] - mn);
  }
  float loss = 0.0f;
  for (int t = 0; t < kT - 1; ++t) {
    if (t + 1 < L) {
      float pc = __expf(y0c[b * (kT - 1) + t] - mc) / sc;
      float pn = __expf(y0n[b * (kT - 1) + t] - mn) / sn;
      loss += -logf(pc) - logf(1.0f - pn);
    }
  }
  atomicAdd(loss_acc, loss);
}

// DIN attention MLP (WMMA): din(32) -> 80 -> 40 -> 1, sigmoid on hidden layers.
__global__ __launch_bounds__(128) void k_att(const float* rnn1, const float* q, const float* W1,
                                             const float* b1, const float* W2, const float* b2,
                                             const float* W3, const float* b3, float* sc) {
  __shared__ _Float16 sW1[32 * 96], sW2[96 * 64], sW3[64 * 16];
  __shared__ _Float16 sIn[4][16 * 32], sH1[4][16 * 96], sH2[4][16 * 64];
  int tid = threadIdx.x, nt = blockDim.x;
  load_weights(W1, 32, 80, sW1, 32, 96, tid, nt);
  load_weights(W2, 80, 40, sW2, 96, 64, tid, nt);
  load_weights(W3, 40, 1, sW3, 64, 16, tid, nt);
  __syncthreads();
  int lane = tid & 31, w = tid >> 5;
  int wave = blockIdx.x * 4 + w, nw = gridDim.x * 4;
  for (int tile = wave; tile < kAttRows / 16; tile += nw) {
    _Float16* in = sIn[w];
    for (int idx = lane; idx < 16 * 32; idx += 32) {
      int row = idx >> 5, col = idx & 31;
      int rg = tile * 16 + row;      // rg == b*T + t
      int b = rg / kT;
      float qv = q[b * 8 + (col & 7)];
      float kv = rnn1[(size_t)rg * 8 + (col & 7)];
      float v = (col < 8) ? qv : (col < 16) ? kv : (col < 24) ? (qv - kv) : (qv * kv);
      in[idx] = (_Float16)v;
    }
    fc_layer<1>(in, 32, sW1, 96, b1, 80, sH1[w], lane);
    fc_layer<1>(sH1[w], 96, sW2, 64, b2, 40, sH2[w], lane);
    v8f acc = wmma_tile(sH2[w], 64, sW3, 16, b3, 1, 0, lane);
    if ((lane & 15) == 0) {
      int rb = (lane >> 4) << 3;
#pragma unroll
      for (int r = 0; r < 8; ++r) sc[tile * 16 + rb + r] = acc[r];
    }
  }
}

// masked softmax over T + VecAttGRU scan, one thread per row
__global__ void k_gru2(const float* rnn1, const float* scv, const int* seqlen, const float* gw,
                       const float* gb, const float* cw, const float* cb, float* hfin) {
  __shared__ float s_gw[16 * 16], s_gb[16], s_cw[16 * 8], s_cb[8];
  int tid = threadIdx.x;
  for (int i = tid; i < 16 * 16; i += blockDim.x) s_gw[i] = gw[i];
  for (int i = tid; i < 16 * 8; i += blockDim.x) s_cw[i] = cw[i];
  for (int i = tid; i < 16; i += blockDim.x) s_gb[i] = gb[i];
  for (int i = tid; i < 8; i += blockDim.x) s_cb[i] = cb[i];
  __syncthreads();
  int b = blockIdx.x * blockDim.x + tid;
  if (b >= kB) return;
  int L = seqlen[b];
  float mx = -1e30f;
  for (int t = 0; t < kT; ++t)
    if (t < L) mx = fmaxf(mx, scv[b * kT + t]);
  float al[kT], sum = 0.0f;
  for (int t = 0; t < kT; ++t) {
    al[t] = (t < L) ? __expf(scv[b * kT + t] - mx) : 0.0f;
    sum += al[t];
  }
  float h[8];
#pragma unroll
  for (int u = 0; u < 8; ++u) h[u] = 0.0f;
  for (int t = 0; t < kT; ++t) {
    if (t >= L) continue;
    float a = al[t] / sum;
    const float* x = rnn1 + ((size_t)(b * kT + t)) * 8;
    float cat[16];
#pragma unroll
    for (int e = 0; e < 8; ++e) { cat[e] = x[e]; cat[8 + e] = h[e]; }
    float g[16];
#pragma unroll
    for (int n = 0; n < 16; ++n) {
      float v = s_gb[n];
#pragma unroll
      for (int k = 0; k < 16; ++k) v += cat[k] * s_gw[k * 16 + n];
      g[n] = sigf(v);
    }
    float cat2[16];
#pragma unroll
    for (int e = 0; e < 8; ++e) { cat2[e] = x[e]; cat2[8 + e] = g[e] * h[e]; }
#pragma unroll
    for (int n = 0; n < 8; ++n) {
      float v = s_cb[n];
#pragma unroll
      for (int k = 0; k < 16; ++k) v += cat2[k] * s_cw[k * 8 + n];
      float c = tanhf(v);
      float u = (1.0f - a) * g[8 + n];
      h[n] = u * h[n] + (1.0f - u) * c;
    }
  }
#pragma unroll
  for (int u = 0; u < 8; ++u) hfin[b * 8 + u] = h[u];
}

// build top features [B,36] + BN stats
__global__ void k_topf(const float* uid_e, const float* item_e, const float* hsum,
                       const float* hfin, float* topf, float* sums, float* sqs) {
  __shared__ float ss[36], sq[36];
  if (threadIdx.x < 36) { ss[threadIdx.x] = 0.0f; sq[threadIdx.x] = 0.0f; }
  __syncthreads();
  int b = blockIdx.x * blockDim.x + threadIdx.x;
  if (b < kB) {
    float f[36];
#pragma unroll
    for (int e = 0; e < 4; ++e) f[e] = uid_e[b * 4 + e];
#pragma unroll
    for (int e = 0; e < 8; ++e) {
      f[4 + e] = item_e[b * 8 + e];
      f[12 + e] = hsum[b * 8 + e];
      f[20 + e] = item_e[b * 8 + e] * hsum[b * 8 + e];
      f[28 + e] = hfin[b * 8 + e];
    }
#pragma unroll
    for (int j = 0; j < 36; ++j) {
      topf[b * 36 + j] = f[j];
      atomicAdd(&ss[j], f[j]);
      atomicAdd(&sq[j], f[j] * f[j]);
    }
  }
  __syncthreads();
  if (threadIdx.x < 36) { atomicAdd(&sums[threadIdx.x], ss[threadIdx.x]); atomicAdd(&sqs[threadIdx.x], sq[threadIdx.x]); }
}

// Top FC part A (WMMA): BN(36) -> 200 relu, write d1 to global
__global__ __launch_bounds__(128) void k_top_a(const float* topf, const float* sums,
                                               const float* sqs, const float* bng,
                                               const float* bnb, const float* W1, const float* b1,
                                               float* d1) {
  __shared__ _Float16 sW1[64 * 208];
  __shared__ _Float16 sIn[4][16 * 64];
  int tid = threadIdx.x, nt = blockDim.x;
  load_weights(W1, 36, 200, sW1, 64, 208, tid, nt);
  __syncthreads();
  int lane = tid & 31, w = tid >> 5;
  int wave = blockIdx.x * 4 + w, nw = gridDim.x * 4;
  const float invN = 1.0f / (float)kB;
  for (int tile = wave; tile < kB / 16; tile += nw) {
    _Float16* in = sIn[w];
    for (int idx = lane; idx < 16 * 64; idx += 32) {
      int row = idx >> 6, col = idx & 63;
      float v = 0.0f;
      if (col < 36) {
        float x = topf[(tile * 16 + row) * 36 + col];
        float m = sums[col] * invN;
        float var = sqs[col] * invN - m * m;
        v = (x - m) * rsqrtf(var + 1e-5f) * bng[col] + bnb[col];
      }
      in[idx] = (_Float16)v;
    }
    for (int n0 = 0; n0 < 208; n0 += 16) {
      v8f acc = wmma_tile(in, 64, sW1, 208, b1, 200, n0, lane);
      int col = n0 + (lane & 15);
      int rb = (lane >> 4) << 3;
      if (col < 200)
#pragma unroll
        for (int r = 0; r < 8; ++r)
          d1[(size_t)(tile * 16 + rb + r) * 200 + col] = fmaxf(acc[r], 0.0f);
    }
  }
}

// Top FC part B (WMMA): 200 -> 80 relu -> (W3@wl folded to single column) -> sigmoid prob
__global__ __launch_bounds__(32) void k_top_b(const float* d1, const float* W2, const float* b2,
                                              const float* W3, const float* b3, const float* wl,
                                              const float* bl, float* prob) {
  __shared__ _Float16 sW2[224 * 96], sW3[96 * 16];
  __shared__ _Float16 sIn[16 * 224], sH2[16 * 96];
  __shared__ float sB3[16];
  int lane = threadIdx.x;
  load_weights(W2, 200, 80, sW2, 224, 96, lane, 32);
  for (int i = lane; i < 96 * 16; i += 32) {
    int k = i / 16, n = i - k * 16;
    float v = (n == 0 && k < 80) ? (W3[k * 2 + 0] * wl[0] + W3[k * 2 + 1] * wl[1]) : 0.0f;
    sW3[i] = (_Float16)v;
  }
  if (lane < 16) sB3[lane] = (lane == 0) ? (b3[0] * wl[0] + b3[1] * wl[1] + bl[0]) : 0.0f;
  __syncthreads();
  for (int tile = blockIdx.x; tile < kB / 16; tile += gridDim.x) {
    int nxt = tile + gridDim.x;
    if (nxt < kB / 16) __builtin_prefetch(d1 + (size_t)nxt * 16 * 200, 0, 1);
    for (int idx = lane; idx < 16 * 224; idx += 32) {
      int row = idx / 224, col = idx - row * 224;
      float v = (col < 200) ? d1[(size_t)(tile * 16 + row) * 200 + col] : 0.0f;
      sIn[idx] = (_Float16)v;
    }
    fc_layer<2>(sIn, 224, sW2, 96, b2, 80, sH2, lane);
    v8f acc = wmma_tile(sH2, 96, sW3, 16, sB3, 1, 0, lane);
    if ((lane & 15) == 0) {
      int rb = (lane >> 4) << 3;
#pragma unroll
      for (int r = 0; r < 8; ++r) prob[tile * 16 + rb + r] = sigf(acc[r]);
    }
  }
}

__global__ void k_finalize(const float* loss_acc, float* out_loss) {
  if (blockIdx.x == 0 && threadIdx.x == 0)
    out_loss[0] = loss_acc[0] / (float)kAuxRows;
}

// ---------------- host launch ----------------
extern "C" void kernel_launch(void* const* d_in, const int* in_sizes, int n_in, void* d_out,
                              int out_size, void* d_ws, size_t ws_size, hipStream_t stream) {
  (void)in_sizes; (void)n_in; (void)out_size; (void)ws_size;
  const int* UID = (const int*)d_in[0];
  const int* ITEM = (const int*)d_in[1];
  const int* CAT = (const int*)d_in[2];
  const int* HI = (const int*)d_in[3];
  const int* HC = (const int*)d_in[4];
  const int* NI = (const int*)d_in[5];
  const int* NC = (const int*)d_in[6];
  const int* SEQ = (const int*)d_in[7];
  const float* emb_uid = (const float*)d_in[8];
  const float* emb_item = (const float*)d_in[9];
  const float* emb_cat = (const float*)d_in[10];
  const float* g1_wih = (const float*)d_in[11];
  const float* g1_whh = (const float*)d_in[12];
  const float* g1_bih = (const float*)d_in[13];
  const float* g1_bhh = (const float*)d_in[14];
  const float* aux_bng = (const float*)d_in[15];
  const float* aux_bnb = (const float*)d_in[16];
  const float* aux_w1 = (const float*)d_in[17];
  const float* aux_b1 = (const float*)d_in[18];
  const float* aux_w2 = (const float*)d_in[19];
  const float* aux_b2 = (const float*)d_in[20];
  const float* aux_w3 = (const float*)d_in[21];
  const float* aux_b3 = (const float*)d_in[22];
  const float* att_qw = (const float*)d_in[23];
  const float* att_qb = (const float*)d_in[24];
  const float* att_prelu = (const float*)d_in[25];
  const float* att_w1 = (const float*)d_in[26];
  const float* att_b1 = (const float*)d_in[27];
  const float* att_w2 = (const float*)d_in[28];
  const float* att_b2 = (const float*)d_in[29];
  const float* att_w3 = (const float*)d_in[30];
  const float* att_b3 = (const float*)d_in[31];
  const float* g2_gw = (const float*)d_in[32];
  const float* g2_gb = (const float*)d_in[33];
  const float* g2_cw = (const float*)d_in[34];
  const float* g2_cb = (const float*)d_in[35];
  const float* top_bng = (const float*)d_in[36];
  const float* top_bnb = (const float*)d_in[37];
  const float* top_w1 = (const float*)d_in[38];
  const float* top_b1 = (const float*)d_in[39];
  const float* top_w2 = (const float*)d_in[40];
  const float* top_b2 = (const float*)d_in[41];
  const float* top_w3 = (const float*)d_in[42];
  const float* top_b3 = (const float*)d_in[43];
  const float* top_wl = (const float*)d_in[44];
  const float* top_bl = (const float*)d_in[45];

  float* out = (float*)d_out;  // [0..B-1] = prob, [B] = aux_loss

  // workspace layout
  float* ws = (float*)d_ws;
  size_t o = 0;
  float* his = ws + o;    o += (size_t)kB * kT * 8;
  float* noclk = ws + o;  o += (size_t)kB * kT * 8;
  float* rnn1 = ws + o;   o += (size_t)kB * kT * 8;
  float* uid_e = ws + o;  o += (size_t)kB * 4;
  float* item_e = ws + o; o += (size_t)kB * 8;
  float* hsum = ws + o;   o += (size_t)kB * 8;
  float* qbuf = ws + o;   o += (size_t)kB * 8;
  float* y0c = ws + o;    o += (size_t)kAuxRows;
  float* y0n = ws + o;    o += (size_t)kAuxRows;
  float* scb = ws + o;    o += (size_t)kB * kT;
  float* hfin = ws + o;   o += (size_t)kB * 8;
  float* topf = ws + o;   o += (size_t)kB * 36;
  float* d1 = ws + o;     o += (size_t)kB * 200;
  float* stats = ws + o;  // sums24[24], sqs24[24], tsum[36], tsq[36], loss[1]
  float* sums24 = stats;
  float* sqs24 = stats + 24;
  float* tsum = stats + 48;
  float* tsq = stats + 84;
  float* loss_acc = stats + 120;

  k_init<<<1, 128, 0, stream>>>(stats, 121);
  k_embed<<<(kB * kT + 255) / 256, 256, 0, stream>>>(HI, HC, NI, NC, emb_item, emb_cat, his, noclk);
  k_embed_b<<<kB / 256, 256, 0, stream>>>(UID, ITEM, CAT, emb_uid, emb_item, emb_cat, his, att_qw,
                                          att_qb, att_prelu, uid_e, item_e, hsum, qbuf);
  k_gru1<<<kB / 256, 256, 0, stream>>>(his, SEQ, g1_wih, g1_whh, g1_bih, g1_bhh, rnn1);
  k_stats24<<<(kAuxRows + 255) / 256, 256, 0, stream>>>(rnn1, his, noclk, sums24, sqs24);
  // click net (features 8-15 -> stats 8..15)
  k_aux<<<1024, 128, 0, stream>>>(rnn1, his, sums24, sqs24, 8, aux_bng, aux_bnb, aux_w1, aux_b1,
                                  aux_w2, aux_b2, aux_w3, aux_b3, y0c);
  // noclick net (features 8-15 -> stats 16..23)
  k_aux<<<1024, 128, 0, stream>>>(rnn1, noclk, sums24, sqs24, 16, aux_bng, aux_bnb, aux_w1, aux_b1,
                                  aux_w2, aux_b2, aux_w3, aux_b3, y0n);
  k_aux_loss<<<kB / 256, 256, 0, stream>>>(y0c, y0n, SEQ, loss_acc);
  k_att<<<1024, 128, 0, stream>>>(rnn1, qbuf, att_w1, att_b1, att_w2, att_b2, att_w3, att_b3, scb);
  k_gru2<<<kB / 256, 256, 0, stream>>>(rnn1, scb, SEQ, g2_gw, g2_gb, g2_cw, g2_cb, hfin);
  k_topf<<<kB / 256, 256, 0, stream>>>(uid_e, item_e, hsum, hfin, topf, tsum, tsq);
  k_top_a<<<128, 128, 0, stream>>>(topf, tsum, tsq, top_bng, top_bnb, top_w1, top_b1, d1);
  k_top_b<<<512, 32, 0, stream>>>(d1, top_w2, top_b2, top_w3, top_b3, top_wl, top_bl, out);
  k_finalize<<<1, 32, 0, stream>>>(loss_acc, out + kB);
}
